// RRPRAMAttention_84112639525098
// MI455X (gfx1250) — matrix operand order repaired and checked by reading, not compiled, not run on hardware
//
#include <hip/hip_runtime.h>
#include <hip/hip_bf16.h>
#include <stdint.h>

typedef __bf16 bf16;
typedef __attribute__((ext_vector_type(16))) __bf16 v16bf;
typedef __attribute__((ext_vector_type(8)))  __bf16 v8bf;
typedef __attribute__((ext_vector_type(4)))  __bf16 v4bf;
typedef __attribute__((ext_vector_type(8)))  float  v8f;
typedef __attribute__((ext_vector_type(4)))  int    v4i;

#if __has_builtin(__builtin_amdgcn_global_load_async_to_lds_b128)
#define USE_ASYNC_LDS 1
#else
#define USE_ASYNC_LDS 0
#endif

__device__ __forceinline__ void wait_async_lds() {
#if USE_ASYNC_LDS
#if __has_builtin(__builtin_amdgcn_s_wait_asynccnt)
  __builtin_amdgcn_s_wait_asynccnt(0);
#else
  asm volatile("s_wait_asynccnt 0x0" ::: "memory");
#endif
#endif
}

// Problem constants (fixed by setup_inputs)
constexpr int Bc  = 4;
constexpr int Tc  = 2048;
constexpr int Ec  = 1024;
constexpr int Hc  = 16;
constexpr int Dc  = 64;
constexpr int HDc = Hc * Dc;   // 1024
constexpr int BHc = Bc * Hc;   // 64
constexpr int CHUNK = 128;     // t-chunk for the scan
constexpr int NC = Tc / CHUNK; // 16

// ---------------------------------------------------------------------------
// f32 -> bf16 bulk convert (one pass; makes both GEMMs stream bf16)
// ---------------------------------------------------------------------------
__global__ __launch_bounds__(256)
void cvt_bf16_kernel(const float* __restrict__ src, bf16* __restrict__ dst, int n4) {
  int i = blockIdx.x * 256 + threadIdx.x;
  if (i < n4) {
    float4 v = *(const float4*)&src[(size_t)i * 4];
    v4bf o;
    o[0] = (bf16)v.x; o[1] = (bf16)v.y; o[2] = (bf16)v.z; o[3] = (bf16)v.w;
    *(v4bf*)&dst[(size_t)i * 4] = o;
  }
}

// ---------------------------------------------------------------------------
// GEMM: C[MxN](f32) = A[MxK](bf16) * B[KxN](bf16), WMMA bf16, f32 accum.
// Workgroup tile 128x128x32, 8 waves, wave tile 32x64 (2x4 WMMA tiles).
// Double-buffered LDS; A staged via GLOBAL_LOAD_ASYNC_TO_LDS (if available),
// B register-staged + transposed into LDS, overlapped with WMMA.
// ---------------------------------------------------------------------------
constexpr int BM = 128, BN = 128, BKt = 32, LPAD = 8;

__global__ __launch_bounds__(256)
void gemm_bf16_wmma(const bf16* __restrict__ A, const bf16* __restrict__ Bm,
                    float* __restrict__ C, int M, int N, int K) {
  __shared__ __align__(16) bf16 As[2][BM][BKt + LPAD];  // [m][k]
  __shared__ __align__(16) bf16 Bs[2][BN][BKt + LPAD];  // transposed [n][k]

  const int tid   = threadIdx.x;
  const int lane  = tid & 31;
  const int wave  = tid >> 5;
  const int waveM = wave >> 1;            // 0..3 -> 32-row slab
  const int waveN = wave & 1;             // 0..1 -> 64-col slab
  const int l15   = lane & 15;
  const int blockM = blockIdx.x * BM;
  const int blockN = blockIdx.y * BN;
  // ISA fragment layout (16-bit A 16x32): lanes 0-15 K 0-7 & 16-23,
  // lanes 16-31 K 8-15 & 24-31. B 32x16: same K split as A.
  const int ak = (lane < 16) ? 0 : 8;
  const int bk = (lane < 16) ? 0 : 8;

  v8f zero = {};
  v8f acc[2][4];
#pragma unroll
  for (int mt = 0; mt < 2; ++mt)
#pragma unroll
    for (int nt = 0; nt < 4; ++nt) acc[mt][nt] = zero;

  v8bf breg[2];

  // --- A tile: 128x32 bf16 = 512 16B chunks, 2 per thread, layout-preserving
  auto stageA = [&](int k0, int buf) {
#pragma unroll
    for (int i = 0; i < 2; ++i) {
      int idx = tid + i * 256;
      int ar = idx >> 2, ac = (idx & 3) << 3;
      const bf16* gp = &A[(size_t)(blockM + ar) * K + k0 + ac];
      bf16* lp = &As[buf][ar][ac];
#if USE_ASYNC_LDS
      __builtin_amdgcn_global_load_async_to_lds_b128((v4i*)gp, (v4i*)lp, 0, 0);
#else
      *(v8bf*)lp = *(const v8bf*)gp;
#endif
    }
  };
  // --- B tile: 32x128 bf16; load 2x v8bf per thread (coalesced along N)...
  auto loadB = [&](int k0) {
#pragma unroll
    for (int i = 0; i < 2; ++i) {
      int idx = tid + i * 256;
      int kr = idx >> 4, nc = (idx & 15) << 3;
      breg[i] = *(const v8bf*)&Bm[(size_t)(k0 + kr) * N + blockN + nc];
    }
  };
  // ...then scatter transposed into LDS (done after WMMA to overlap loads)
  auto storeB = [&](int buf) {
#pragma unroll
    for (int i = 0; i < 2; ++i) {
      int idx = tid + i * 256;
      int kr = idx >> 4, nc = (idx & 15) << 3;
#pragma unroll
      for (int j = 0; j < 8; ++j) Bs[buf][nc + j][kr] = breg[i][j];
    }
  };

  auto compute = [&](int buf) {
#pragma unroll
    for (int mt = 0; mt < 2; ++mt) {
      const bf16* ap = &As[buf][waveM * 32 + mt * 16 + l15][ak];
      v8bf a0 = *(const v8bf*)ap;            // K ak..ak+7
      v8bf a1 = *(const v8bf*)(ap + 16);     // K ak+16..ak+23
      v16bf a;
#pragma unroll
      for (int i = 0; i < 8; ++i) { a[i] = a0[i]; a[i + 8] = a1[i]; }
#pragma unroll
      for (int nt = 0; nt < 4; ++nt) {
        const bf16* bp = &Bs[buf][waveN * 64 + nt * 16 + l15][bk];
        v8bf b0 = *(const v8bf*)bp;          // K bk..bk+7
        v8bf b1 = *(const v8bf*)(bp + 16);   // K bk+16..bk+23
        v16bf b;
#pragma unroll
        for (int i = 0; i < 8; ++i) { b[i] = b0[i]; b[i + 8] = b1[i]; }
        acc[mt][nt] = __builtin_amdgcn_wmma_f32_16x16x32_bf16(
            false, a, false, b, (short)0, acc[mt][nt], false, false);
      }
    }
  };

  const int nk = K / BKt;
  // prologue: stage tile 0
  loadB(0);
  stageA(0, 0);
  storeB(0);
  wait_async_lds();
  __syncthreads();

  for (int kt = 0; kt < nk; ++kt) {
    const int cur = kt & 1, nxt = cur ^ 1;
    const bool more = (kt + 1 < nk);
    if (more) {
      loadB((kt + 1) * BKt);          // B -> registers (in flight during WMMA)
      stageA((kt + 1) * BKt, nxt);    // A -> LDS, async, other buffer
      if (kt + 2 < nk)                // hint tile kt+2 into L2
        __builtin_prefetch(&A[(size_t)(blockM + (tid >> 2)) * K + (kt + 2) * BKt], 0, 0);
    }
    compute(cur);
    if (more) storeB(nxt);
    wait_async_lds();
    __syncthreads();
  }

  // C/D layout: VGPR r: lanes 0-15 -> M=r, N=lane; lanes 16-31 -> M=8+r, N=lane-16.
  const int mrow = blockM + waveM * 32 + ((lane < 16) ? 0 : 8);
#pragma unroll
  for (int mt = 0; mt < 2; ++mt)
#pragma unroll
    for (int nt = 0; nt < 4; ++nt) {
      int n = blockN + waveN * 64 + nt * 16 + l15;
#pragma unroll
      for (int r = 0; r < 8; ++r)
        C[(size_t)(mrow + mt * 16 + r) * N + n] = acc[mt][nt][r];
    }
}

// ---------------------------------------------------------------------------
// Scores: s[b,h,t] = (1/8) * sum_e x[b,t,e] * wr[h,e,t].
// One block per 32-t tile handles all b,h -> wr and x each read exactly once.
// ---------------------------------------------------------------------------
constexpr int ST = 32, SEc = 128;

__global__ __launch_bounds__(256)
void scores_kernel(const float* __restrict__ x, const float* __restrict__ wr,
                   float* __restrict__ s) {
  __shared__ float xs[Bc][ST][SEc];   // 64 KB
  const int tid = threadIdx.x;
  const int t0  = blockIdx.x * ST;
  const int tl  = tid & 31;
  const int h0  = tid >> 5;           // 0..7; thread also covers h0+8
  float acc[2][Bc];
#pragma unroll
  for (int i = 0; i < 2; ++i)
#pragma unroll
    for (int b = 0; b < Bc; ++b) acc[i][b] = 0.f;

  for (int e0 = 0; e0 < Ec; e0 += SEc) {
    __syncthreads();
#pragma unroll
    for (int i = 0; i < 16; ++i) {
      int idx  = tid + i * 256;
      int e4   = (idx & 31) << 2;
      int rest = idx >> 5;
      int b = rest >> 5, tt = rest & 31;
      *(float4*)&xs[b][tt][e4] =
          *(const float4*)&x[((size_t)b * Tc + t0 + tt) * Ec + e0 + e4];
    }
    __syncthreads();
#pragma unroll
    for (int hh = 0; hh < 2; ++hh) {
      int h = h0 + hh * 8;
      for (int e = 0; e < SEc; ++e) {
        float w = wr[((size_t)h * Ec + e0 + e) * Tc + t0 + tl];  // coalesced in t
#pragma unroll
        for (int b = 0; b < Bc; ++b) acc[hh][b] += xs[b][tl][e] * w;
      }
    }
  }
#pragma unroll
  for (int hh = 0; hh < 2; ++hh)
#pragma unroll
    for (int b = 0; b < Bc; ++b)
      s[(size_t)(b * Hc + h0 + hh * 8) * Tc + t0 + tl] = acc[hh][b] * 0.125f;
}

// ---------------------------------------------------------------------------
// Per (b,h): global max, e = exp(s-m), den = inclusive cumsum(e).
// ---------------------------------------------------------------------------
__global__ __launch_bounds__(256)
void softmax_scan_kernel(const float* __restrict__ s, float* __restrict__ eb,
                         float* __restrict__ den) {
  const int bh = blockIdx.x;
  const float* sp = s + (size_t)bh * Tc;
  __shared__ float red[256];
  const int tid = threadIdx.x;
  float vals[8];
  float mx = -3.4e38f;
#pragma unroll
  for (int i = 0; i < 8; ++i) { vals[i] = sp[tid * 8 + i]; mx = fmaxf(mx, vals[i]); }
  red[tid] = mx; __syncthreads();
  for (int off = 128; off > 0; off >>= 1) {
    if (tid < off) red[tid] = fmaxf(red[tid], red[tid + off]);
    __syncthreads();
  }
  const float m = red[0];
  __syncthreads();
  float sum = 0.f;
#pragma unroll
  for (int i = 0; i < 8; ++i) { vals[i] = __expf(vals[i] - m); sum += vals[i]; }
  red[tid] = sum; __syncthreads();
  for (int off = 1; off < 256; off <<= 1) {    // Hillis-Steele inclusive scan
    float v = red[tid];
    float add = (tid >= off) ? red[tid - off] : 0.f;
    __syncthreads();
    red[tid] = v + add;
    __syncthreads();
  }
  float run = red[tid] - sum;                  // exclusive prefix for this thread
  float* ep = eb  + (size_t)bh * Tc;
  float* dp = den + (size_t)bh * Tc;
#pragma unroll
  for (int i = 0; i < 8; ++i) {
    run += vals[i];
    ep[tid * 8 + i] = vals[i];
    dp[tid * 8 + i] = run;
  }
}

// ---------------------------------------------------------------------------
// Chunked weighted cumsum over V:  P[bh,c,d] = sum_{t in chunk c} e_t * v[t,d]
// then att[b,t,h*D+d] = (prefix(c) + running) / den_t, emitted directly in bf16.
// v/att are row-major [B*T, H*D] (GEMM layout).
// ---------------------------------------------------------------------------
__global__ __launch_bounds__(64)
void chunk_partial_kernel(const float* __restrict__ eb, const float* __restrict__ v,
                          float* __restrict__ P) {
  const int bh = blockIdx.x, c = blockIdx.y, d = threadIdx.x;
  const int b = bh >> 4, h = bh & 15;
  const float* ep = eb + (size_t)bh * Tc + c * CHUNK;
  float acc = 0.f;
  for (int i = 0; i < CHUNK; ++i) {
    int t = c * CHUNK + i;
    acc += ep[i] * v[(size_t)(b * Tc + t) * HDc + h * Dc + d];
  }
  P[((size_t)bh * NC + c) * Dc + d] = acc;
}

__global__ __launch_bounds__(64)
void chunk_scan_kernel(const float* __restrict__ eb, const float* __restrict__ den,
                       const float* __restrict__ v, const float* __restrict__ P,
                       bf16* __restrict__ att) {
  const int bh = blockIdx.x, c = blockIdx.y, d = threadIdx.x;
  const int b = bh >> 4, h = bh & 15;
  float acc = 0.f;
  for (int cc = 0; cc < c; ++cc) acc += P[((size_t)bh * NC + cc) * Dc + d];
  for (int i = 0; i < CHUNK; ++i) {
    int t = c * CHUNK + i;
    size_t vi = (size_t)(b * Tc + t) * HDc + h * Dc + d;
    acc += eb[(size_t)bh * Tc + t] * v[vi];
    att[vi] = (bf16)(acc / den[(size_t)bh * Tc + t]);
  }
}

// ---------------------------------------------------------------------------
extern "C" void kernel_launch(void* const* d_in, const int* in_sizes, int n_in,
                              void* d_out, int out_size, void* d_ws, size_t ws_size,
                              hipStream_t stream) {
  (void)in_sizes; (void)n_in; (void)out_size; (void)ws_size;
  const float* x  = (const float*)d_in[0];   // [B,T,E]
  const float* wv = (const float*)d_in[1];   // [E,H*D]
  const float* wr = (const float*)d_in[2];   // [H,E,T]
  const float* wo = (const float*)d_in[3];   // [H*D,E]
  float* out = (float*)d_out;                // [B,T,E]

  char* p = (char*)d_ws;
  auto carve = [&](size_t bytes) { char* r = p; p += (bytes + 255) & ~(size_t)255; return r; };
  float* s_buf   = (float*)carve((size_t)BHc * Tc * 4);
  float* e_buf   = (float*)carve((size_t)BHc * Tc * 4);
  float* den_buf = (float*)carve((size_t)BHc * Tc * 4);
  float* P_buf   = (float*)carve((size_t)BHc * NC * Dc * 4);
  float* v_buf   = (float*)carve((size_t)Bc * Tc * HDc * 4);
  bf16*  xb      = (bf16*)carve((size_t)Bc * Tc * Ec * 2);
  bf16*  wvb     = (bf16*)carve((size_t)Ec * HDc * 2);
  bf16*  wob     = (bf16*)carve((size_t)HDc * Ec * 2);
  bf16*  attb    = (bf16*)carve((size_t)Bc * Tc * HDc * 2);

  // 0) one-time f32 -> bf16 conversions for all GEMM operands
  int nx4 = Bc * Tc * Ec / 4, nw4 = Ec * HDc / 4;
  cvt_bf16_kernel<<<(nx4 + 255) / 256, 256, 0, stream>>>(x,  xb,  nx4);
  cvt_bf16_kernel<<<(nw4 + 255) / 256, 256, 0, stream>>>(wv, wvb, nw4);
  cvt_bf16_kernel<<<(nw4 + 255) / 256, 256, 0, stream>>>(wo, wob, nw4);

  // 1) v = x @ wv   : [8192,1024] x [1024,1024]
  dim3 gv(Bc * Tc / BM, HDc / BN);
  gemm_bf16_wmma<<<gv, 256, 0, stream>>>(xb, wvb, v_buf, Bc * Tc, HDc, Ec);

  // 2) scores s[b,h,t]
  scores_kernel<<<Tc / ST, 256, 0, stream>>>(x, wr, s_buf);

  // 3) softmax numerators e and cumulative denominators
  softmax_scan_kernel<<<BHc, 256, 0, stream>>>(s_buf, e_buf, den_buf);

  // 4/5) chunked prefix-weighted cumsum over V -> att (bf16)
  chunk_partial_kernel<<<dim3(BHc, NC), Dc, 0, stream>>>(e_buf, v_buf, P_buf);
  chunk_scan_kernel<<<dim3(BHc, NC), Dc, 0, stream>>>(e_buf, den_buf, v_buf, P_buf, attb);

  // 6) out = att @ wo : [8192,1024] x [1024,1024]
  dim3 go(Bc * Tc / BM, Ec / BN);
  gemm_bf16_wmma<<<go, 256, 0, stream>>>(attb, wob, out, Bc * Tc, Ec, HDc);
}
